// MultiHeadAttentionBlock_18897856102984
// MI455X (gfx1250) — matrix operand order repaired and checked by reading, not compiled, not run on hardware
//
#include <hip/hip_runtime.h>
#include <hip/hip_bf16.h>
#include <stdint.h>

// ---------------------------------------------------------------------------
// MHA block, CDNA5 (gfx1250), wave32, bf16 WMMA everywhere.
// B=4, S=2048, D=1024, H=16, DK=64
// ---------------------------------------------------------------------------

#define B_  4
#define S_  2048
#define D_  1024
#define H_  16
#define DK_ 64

typedef __bf16 bf16;
typedef __attribute__((ext_vector_type(16))) __bf16 v16bf;
typedef __attribute__((ext_vector_type(8)))  __bf16 bf16x8;
typedef __attribute__((ext_vector_type(8)))  float  v8f;
typedef __attribute__((ext_vector_type(4)))  float  f32x4;
typedef int i32x4 __attribute__((vector_size(16)));   // matches builtin param

// ---- CDNA5 async global->LDS path (guarded; falls back to sync copy) ------
#if defined(__has_builtin)
#  if __has_builtin(__builtin_amdgcn_global_load_async_to_lds_b128)
#    define USE_ASYNC_LDS 1
#  endif
#  if __has_builtin(__builtin_amdgcn_s_wait_asynccnt)
#    define HAS_WAIT_ASYNC 1
#  endif
#endif
#ifndef USE_ASYNC_LDS
#  define USE_ASYNC_LDS 0
#endif
#ifndef HAS_WAIT_ASYNC
#  define HAS_WAIT_ASYNC 0
#endif

// copy 16 bytes global->LDS at byte offset OFF (imm) from both base pointers
template <int OFF>
__device__ __forceinline__ void stage16(const bf16* g, bf16* l) {
#if USE_ASYNC_LDS
  __builtin_amdgcn_global_load_async_to_lds_b128(
      (i32x4*)const_cast<bf16*>(g), (i32x4*)l, OFF, 0);
#else
  *(bf16x8*)((char*)l + OFF) = *(const bf16x8*)((const char*)g + OFF);
#endif
}

__device__ __forceinline__ void wait_async() {
#if USE_ASYNC_LDS
#  if HAS_WAIT_ASYNC
  __builtin_amdgcn_s_wait_asynccnt(0);
#  else
  asm volatile("s_wait_asynccnt 0" ::: "memory");
#  endif
#endif
}

__device__ __forceinline__ bf16 f2bf(float f) {
  union { float f; uint32_t u; } x; x.f = f;
  uint32_t r = x.u + 0x7FFFu + ((x.u >> 16) & 1u);   // round-to-nearest-even
  uint16_t h = (uint16_t)(r >> 16);
  bf16 b; __builtin_memcpy(&b, &h, 2);
  return b;
}

__device__ __forceinline__ v16bf concat8(bf16x8 lo, bf16x8 hi) {
  return __builtin_shufflevector(lo, hi, 0,1,2,3,4,5,6,7,8,9,10,11,12,13,14,15);
}

__device__ __forceinline__ v8f wmma_bf16(v16bf a, v16bf b, v8f c) {
  // D = A(16x32) * B(32x16) + C, f32 accumulate
  return __builtin_amdgcn_wmma_f32_16x16x32_bf16(false, a, false, b, (short)0, c,
                                                 false, false);
}

// ---------------------------------------------------------------------------
// f32 -> bf16 elementwise (weights)
// ---------------------------------------------------------------------------
__global__ void cvt_f32_bf16(const float* __restrict__ src,
                             bf16* __restrict__ dst, int n) {
  int i = blockIdx.x * blockDim.x + threadIdx.x;
  if (i < n) dst[i] = f2bf(src[i]);
}

// ---------------------------------------------------------------------------
// proj_gemm: Y = X (f32, M x K, row-major) @ W^T, W is (N x K) bf16 row-major.
// M = B*S = 8192, K = N = 1024. Workgroup: 256 thr (8 waves), tile 128(M)x64(N).
// Each wave: 16(M) x 64(N) via 4 WMMA accumulators; K staged 64-wide in LDS.
// MODE 0: Q -> (BH,S,DK) scaled by 1/8   (softmax scale folded in)
// MODE 1: K -> (BH,S,DK)
// MODE 2: V -> (BH,DK,S)  (transposed so PV GEMM B-operand rows are contiguous)
// ---------------------------------------------------------------------------
template <int MODE>
__global__ __launch_bounds__(256)
void proj_gemm(const float* __restrict__ X, const bf16* __restrict__ W,
               bf16* __restrict__ Y) {
  const int K    = D_;
  const int m0   = blockIdx.x * 128;
  const int n0   = blockIdx.y * 64;
  const int tid  = threadIdx.x;
  const int lane = tid & 31, wave = tid >> 5;
  const int lq   = lane & 15, half = lane >> 4;
  const int mw   = m0 + wave * 16;
  const int nrow = tid >> 2, p = tid & 3;      // staging coords: 64 rows x 4x16

  __shared__ bf16 Wlds[64][72];   // 64(N) x 64(K) slab, +8 pad (stride 36 dw)

  v8f acc[4] = {};

  for (int k0 = 0; k0 < K; k0 += 64) {
    __syncthreads();
    {   // stage W slab: 64 rows x 64 K = 8KB, 32B per thread (async if avail)
      const bf16* wsrc = W + (size_t)(n0 + nrow) * K + k0 + p * 16;
      bf16* wdst = &Wlds[nrow][p * 16];
      stage16<0>(wsrc, wdst);
      stage16<16>(wsrc, wdst);
      wait_async();
    }
    __syncthreads();
    if (k0 + 64 < K)   // prefetch next W slab into cache
      __builtin_prefetch(W + (size_t)(n0 + nrow) * K + k0 + 64 + p * 16, 0, 3);

#pragma unroll
    for (int ks = 0; ks < 2; ++ks) {
      const int kk = k0 + ks * 32;
      // A fragment: row mw+lq, K chunks {kb..kb+7},{kb+16..kb+23}, kb=kk+half*8
      const float* ap = X + (size_t)(mw + lq) * K + kk + half * 8;
      f32x4 a0 = *(const f32x4*)(ap);
      f32x4 a1 = *(const f32x4*)(ap + 4);
      f32x4 a2 = *(const f32x4*)(ap + 16);
      f32x4 a3 = *(const f32x4*)(ap + 20);
      v16bf afrag;
#pragma unroll
      for (int i = 0; i < 4; ++i) {
        afrag[i]      = f2bf(a0[i]);
        afrag[4 + i]  = f2bf(a1[i]);
        afrag[8 + i]  = f2bf(a2[i]);
        afrag[12 + i] = f2bf(a3[i]);
      }
#pragma unroll
      for (int t = 0; t < 4; ++t) {
        const bf16x8* bp = (const bf16x8*)&Wlds[t * 16 + lq][ks * 32 + half * 8];
        acc[t] = wmma_bf16(afrag, concat8(bp[0], bp[2]), acc[t]);
      }
    }
  }

  // store: lane (half,lq), vgpr r -> row mw+half*8+r, col n0+t*16+lq
#pragma unroll
  for (int t = 0; t < 4; ++t) {
#pragma unroll
    for (int r = 0; r < 8; ++r) {
      int m = mw + half * 8 + r;
      int n = n0 + t * 16 + lq;
      int b = m >> 11, s = m & (S_ - 1);
      int h = n >> 6,  dk = n & 63;
      float val = acc[t][r];
      if (MODE == 0) val *= 0.125f;          // 1/sqrt(DK)
      size_t idx;
      if (MODE == 2) idx = ((size_t)(b * H_ + h) * DK_ + dk) * S_ + s;
      else           idx = ((size_t)(b * H_ + h) * S_ + s) * DK_ + dk;
      Y[idx] = f2bf(val);
    }
  }
}

// ---------------------------------------------------------------------------
// Flash attention. Grid: (BH=64, S/128=16). WG = 256 thr (8 waves).
// Wave w owns 16 query rows [q0+16w, q0+16w+16). Key blocks of 64, causal.
// Q pre-scaled by 1/8. Qh,Kh: (BH,S,64) bf16. Vt: (BH,64,S) bf16.
// Output Og: (B,S,D) bf16.
// ---------------------------------------------------------------------------
__global__ __launch_bounds__(256)
void attn_flash(const bf16* __restrict__ Qh, const bf16* __restrict__ Kh,
                const bf16* __restrict__ Vt, bf16* __restrict__ Og) {
  const int bh = blockIdx.x;
  const int q0 = blockIdx.y * 128;
  const int b  = bh >> 4, h = bh & 15;
  const int tid  = threadIdx.x;
  const int lane = tid & 31, wave = tid >> 5;
  const int lq   = lane & 15, half = lane >> 4;
  const int mw   = q0 + wave * 16;
  const int rrow = tid >> 2, p = tid & 3;    // staging coords

  __shared__ bf16 Klds[64][72];        // 64 keys x 64 dk, +8 pad
  __shared__ bf16 Vlds[64][72];        // 64 dk x 64 keys (transposed V), +8 pad
  __shared__ bf16 Plds[8][16][72];     // per-wave P tile 16 x 64, +8 pad

  // Q A-fragments (two K-steps over DK=64), loaded once
  const bf16* Qbase = Qh + ((size_t)bh * S_ + (mw + lq)) * DK_;
  v16bf qf[2];
#pragma unroll
  for (int ks = 0; ks < 2; ++ks) {
    int kb = ks * 32 + half * 8;
    bf16x8 a0 = *(const bf16x8*)(Qbase + kb);
    bf16x8 a1 = *(const bf16x8*)(Qbase + kb + 16);
    qf[ks] = concat8(a0, a1);
  }

  v8f o[4] = {};
  float mrow[8], lrow[8];
#pragma unroll
  for (int r = 0; r < 8; ++r) { mrow[r] = -1e30f; lrow[r] = 0.f; }

  const bf16* Kb = Kh + (size_t)bh * S_ * DK_;
  const bf16* Vb = Vt + (size_t)bh * DK_ * S_;
  const int nkb = (q0 >> 6) + 2;   // causal: key blocks up to q0+128

  for (int ib = 0; ib < nkb; ++ib) {
    const int kb = ib * 64;
    __syncthreads();
    {   // stage K block (8KB) + V block (8KB): 64B per thread, async if avail
      const bf16* kr = Kb + (size_t)(kb + rrow) * DK_ + p * 16;
      bf16* kl = &Klds[rrow][p * 16];
      stage16<0>(kr, kl);
      stage16<16>(kr, kl);
      const bf16* vr = Vb + (size_t)rrow * S_ + kb + p * 16;
      bf16* vl = &Vlds[rrow][p * 16];
      stage16<0>(vr, vl);
      stage16<16>(vr, vl);
      wait_async();
    }
    __syncthreads();
    if (ib + 1 < nkb) {   // prefetch next K/V block while this one computes
      __builtin_prefetch(Kb + (size_t)(kb + 64 + rrow) * DK_ + p * 16, 0, 3);
      __builtin_prefetch(Vb + (size_t)rrow * S_ + kb + 64 + p * 16, 0, 3);
    }

    if (kb > mw + 15) continue;   // block fully above diagonal for this wave

    // scores: S = Q (16x64) @ K_block^T (64x64) -> 4 tiles of 16x16
    v8f sc[4];
#pragma unroll
    for (int t = 0; t < 4; ++t) {
      const bf16x8* bp0 = (const bf16x8*)&Klds[t * 16 + lq][half * 8];
      const bf16x8* bp1 = (const bf16x8*)&Klds[t * 16 + lq][32 + half * 8];
      v8f c = {};
      c = wmma_bf16(qf[0], concat8(bp0[0], bp0[2]), c);
      c = wmma_bf16(qf[1], concat8(bp1[0], bp1[2]), c);
      sc[t] = c;
    }

    // causal mask (only when this wave's tile can touch the diagonal)
    const int rowbase = mw + half * 8;
    if (kb + 63 > mw) {
#pragma unroll
      for (int t = 0; t < 4; ++t) {
        int col = kb + t * 16 + lq;
#pragma unroll
        for (int r = 0; r < 8; ++r)
          if (col > rowbase + r) sc[t][r] = -1e30f;
      }
    }

    // online softmax: row max across 64 cols (4 tiles x 16-lane halves)
    float mnew[8];
#pragma unroll
    for (int r = 0; r < 8; ++r) {
      float v = fmaxf(fmaxf(sc[0][r], sc[1][r]), fmaxf(sc[2][r], sc[3][r]));
#pragma unroll
      for (int off = 8; off >= 1; off >>= 1)
        v = fmaxf(v, __shfl_xor(v, off, 32));
      mnew[r] = fmaxf(mrow[r], v);
    }
#pragma unroll
    for (int r = 0; r < 8; ++r) {
      float corr = __expf(mrow[r] - mnew[r]);
      lrow[r] *= corr;
#pragma unroll
      for (int t = 0; t < 4; ++t) o[t][r] *= corr;
      mrow[r] = mnew[r];
    }
#pragma unroll
    for (int t = 0; t < 4; ++t)
#pragma unroll
      for (int r = 0; r < 8; ++r)
        sc[t][r] = __expf(sc[t][r] - mrow[r]);
#pragma unroll
    for (int r = 0; r < 8; ++r) {
      float v = sc[0][r] + sc[1][r] + sc[2][r] + sc[3][r];
#pragma unroll
      for (int off = 8; off >= 1; off >>= 1)
        v += __shfl_xor(v, off, 32);
      lrow[r] += v;
    }

    // P tile C-layout -> A-layout via per-wave LDS (in-order DS, no barrier)
#pragma unroll
    for (int t = 0; t < 4; ++t)
#pragma unroll
      for (int r = 0; r < 8; ++r)
        Plds[wave][half * 8 + r][t * 16 + lq] = f2bf(sc[t][r]);

    // O += P (16x64) @ V_block (64x64); B-operand rows contiguous in Vlds
#pragma unroll
    for (int ks = 0; ks < 2; ++ks) {
      int kbp = ks * 32 + half * 8;
      const bf16* pr = &Plds[wave][lq][kbp];
      v16bf af = concat8(*(const bf16x8*)pr, *(const bf16x8*)(pr + 16));
#pragma unroll
      for (int t = 0; t < 4; ++t) {
        const bf16x8* vp = (const bf16x8*)&Vlds[t * 16 + lq][ks * 32 + half * 8];
        o[t] = wmma_bf16(af, concat8(vp[0], vp[2]), o[t]);
      }
    }
  }

  // normalize and store to (B,S,D) bf16
#pragma unroll
  for (int r = 0; r < 8; ++r) {
    float inv = 1.0f / lrow[r];
#pragma unroll
    for (int t = 0; t < 4; ++t) o[t][r] *= inv;
  }
#pragma unroll
  for (int t = 0; t < 4; ++t)
#pragma unroll
    for (int r = 0; r < 8; ++r) {
      int srow = mw + half * 8 + r;
      Og[((size_t)b * S_ + srow) * D_ + h * 64 + t * 16 + lq] = f2bf(o[t][r]);
    }
}

// ---------------------------------------------------------------------------
// out_gemm: out(f32, M x N) = X(bf16, M x K) @ Wo^T, Wo (N x K) bf16.
// Same tiling as proj_gemm; stores f32 directly to d_out.
// ---------------------------------------------------------------------------
__global__ __launch_bounds__(256)
void out_gemm(const bf16* __restrict__ X, const bf16* __restrict__ W,
              float* __restrict__ Y) {
  const int K    = D_;
  const int m0   = blockIdx.x * 128;
  const int n0   = blockIdx.y * 64;
  const int tid  = threadIdx.x;
  const int lane = tid & 31, wave = tid >> 5;
  const int lq   = lane & 15, half = lane >> 4;
  const int mw   = m0 + wave * 16;
  const int nrow = tid >> 2, p = tid & 3;

  __shared__ bf16 Wlds[64][72];

  v8f acc[4] = {};

  for (int k0 = 0; k0 < K; k0 += 64) {
    __syncthreads();
    {
      const bf16* wsrc = W + (size_t)(n0 + nrow) * K + k0 + p * 16;
      bf16* wdst = &Wlds[nrow][p * 16];
      stage16<0>(wsrc, wdst);
      stage16<16>(wsrc, wdst);
      wait_async();
    }
    __syncthreads();
    if (k0 + 64 < K)
      __builtin_prefetch(W + (size_t)(n0 + nrow) * K + k0 + 64 + p * 16, 0, 3);

#pragma unroll
    for (int ks = 0; ks < 2; ++ks) {
      const int kk = k0 + ks * 32;
      const bf16* ap = X + (size_t)(mw + lq) * K + kk + half * 8;
      v16bf afrag = concat8(*(const bf16x8*)ap, *(const bf16x8*)(ap + 16));
#pragma unroll
      for (int t = 0; t < 4; ++t) {
        const bf16x8* bp = (const bf16x8*)&Wlds[t * 16 + lq][ks * 32 + half * 8];
        acc[t] = wmma_bf16(afrag, concat8(bp[0], bp[2]), acc[t]);
      }
    }
  }

#pragma unroll
  for (int t = 0; t < 4; ++t)
#pragma unroll
    for (int r = 0; r < 8; ++r) {
      int m = mw + half * 8 + r;
      int n = n0 + t * 16 + lq;
      Y[(size_t)m * D_ + n] = acc[t][r];
    }
}

// ---------------------------------------------------------------------------
extern "C" void kernel_launch(void* const* d_in, const int* in_sizes, int n_in,
                              void* d_out, int out_size, void* d_ws, size_t ws_size,
                              hipStream_t stream) {
  (void)in_sizes; (void)n_in; (void)out_size; (void)ws_size;
  const float* q  = (const float*)d_in[0];
  const float* k  = (const float*)d_in[1];
  const float* v  = (const float*)d_in[2];
  // d_in[3] = mask (causal, hardcoded in kernel)
  const float* Wq = (const float*)d_in[4];
  const float* Wk = (const float*)d_in[5];
  const float* Wv = (const float*)d_in[6];
  const float* Wo = (const float*)d_in[7];
  float* out = (float*)d_out;

  const size_t DD = (size_t)D_ * D_;        // 1M
  const size_t MD = (size_t)B_ * S_ * D_;   // 8M

  bf16* Wq_b = (bf16*)d_ws;
  bf16* Wk_b = Wq_b + DD;
  bf16* Wv_b = Wk_b + DD;
  bf16* Wo_b = Wv_b + DD;
  bf16* Qh   = Wo_b + DD;
  bf16* Kh   = Qh + MD;
  bf16* Vt   = Kh + MD;
  bf16* Og   = Vt + MD;   // total ws: 4*DD + 4*MD bf16 = ~75.5 MB

  // 1) weights -> bf16
  int nw = (int)DD;
  int gw = (nw + 255) / 256;
  cvt_f32_bf16<<<gw, 256, 0, stream>>>(Wq, Wq_b, nw);
  cvt_f32_bf16<<<gw, 256, 0, stream>>>(Wk, Wk_b, nw);
  cvt_f32_bf16<<<gw, 256, 0, stream>>>(Wv, Wv_b, nw);
  cvt_f32_bf16<<<gw, 256, 0, stream>>>(Wo, Wo_b, nw);

  // 2) Q/K/V projections (M=8192 -> 64 tiles, N=1024 -> 16 tiles)
  dim3 gg(64, 16);
  proj_gemm<0><<<gg, 256, 0, stream>>>(q, Wq_b, Qh);
  proj_gemm<1><<<gg, 256, 0, stream>>>(k, Wk_b, Kh);
  proj_gemm<2><<<gg, 256, 0, stream>>>(v, Wv_b, Vt);

  // 3) flash attention: (BH=64) x (S/128=16)
  attn_flash<<<dim3(64, 16), 256, 0, stream>>>(Qh, Kh, Vt, Og);

  // 4) output projection -> f32 out
  out_gemm<<<gg, 256, 0, stream>>>(Og, Wo_b, out);
}
